// Attention_21174188769848
// MI455X (gfx1250) — compile-verified, matrix operand order checked by reading, not run on hardware
//
#include <hip/hip_runtime.h>
#include <hip/hip_bf16.h>

// ---------- CDNA5 WMMA types ----------
typedef __attribute__((ext_vector_type(16))) __bf16 v16bf;
typedef __attribute__((ext_vector_type(8)))  float  v8f;

// Async global->LDS path (gfx1250 GLOBAL_LOAD_ASYNC_TO_LDS_*, ASYNCcnt)
#if defined(__has_builtin)
#  if __has_builtin(__builtin_amdgcn_global_load_async_to_lds_b128) && \
      __has_builtin(__builtin_amdgcn_s_wait_asynccnt)
#    define USE_ASYNC_LDS 1
#  else
#    define USE_ASYNC_LDS 0
#  endif
#else
#  define USE_ASYNC_LDS 0
#endif

#if USE_ASYNC_LDS
// Builtin signature (from compiler diagnostic): pointers to int4 with explicit
// address spaces: (global int4*, lds int4*, imm offset, imm cpol)
typedef int b128_t __attribute__((vector_size(16)));
typedef __attribute__((address_space(1))) b128_t GB128;
typedef __attribute__((address_space(3))) b128_t LB128;
#endif

__device__ __forceinline__ void stage_wait() {
#if USE_ASYNC_LDS
    __builtin_amdgcn_s_wait_asynccnt(0);   // own wave's async LDS writes done
#endif
}

__device__ __forceinline__ v8f wmma_bf16(v16bf a, v16bf b, v8f c) {
    // D = A(16x32 bf16) * B(32x16 bf16) + C(16x16 f32)
    return __builtin_amdgcn_wmma_f32_16x16x32_bf16(
        false, a, false, b, (short)0, c, false, false);
}

// float -> bf16 round-to-nearest-even
__device__ __forceinline__ unsigned short f2bf(float x) {
    unsigned int u = __float_as_uint(x);
    u += 0x7FFFu + ((u >> 16) & 1u);
    return (unsigned short)(u >> 16);
}

// ---------- fragment loader ----------
// Tile is row-major [rows][ldk] bf16 (ushort). Loads a 16x32 A-operand
// (or the B-operand of a K^T-oriented tile, i.e. tile stored [N][K]).
// ISA layout (16-bit A 16x32): lane 0-15: row=lane, VGPR j -> K pair;
// lanes 16-31 shift K by +8.  K(pair j) = (j<4?0:16) + 8*(lane/16) + 2*(j%4).
__device__ __forceinline__ v16bf load_frag(const unsigned short* tile,
                                           int row, int k0, int ldk) {
    const int lane  = threadIdx.x & 31;
    const int half8 = (lane >> 4) << 3;  // 0 or 8
    const unsigned short* p = tile + (size_t)(row + (lane & 15)) * ldk + k0 + half8;
    union { v16bf v; unsigned int u[8]; } f;
#pragma unroll
    for (int j = 0; j < 8; ++j) {
        const int kk = ((j & 4) << 2) + ((j & 3) << 1);  // 0,2,4,6,16,18,20,22
        f.u[j] = *(const unsigned int*)(p + kk);
    }
    return f.v;
}

// ---------- LDS staging: copy rows x 64 bf16, 256 threads ----------
__device__ __forceinline__ void stage_tile64(unsigned short* __restrict__ dst,
                                             const unsigned short* __restrict__ src,
                                             int rows, int ldg) {
    const int total = rows * 8;  // uint4 units (8 ushorts each)
    for (int idx = threadIdx.x; idx < total; idx += 256) {
        const int r = idx >> 3, c = idx & 7;
#if USE_ASYNC_LDS
        __builtin_amdgcn_global_load_async_to_lds_b128(
            (GB128*)(src + (size_t)r * ldg + c * 8),
            (LB128*)(dst + r * 64 + c * 8), 0, 0);
#else
        *(uint4*)(dst + r * 64 + c * 8) = *(const uint4*)(src + (size_t)r * ldg + c * 8);
#endif
    }
}

// V tile [64 keys][64 hd] -> LDS transposed [64 hd][64 keys] (B-operand orientation)
__device__ __forceinline__ void stage_vT(unsigned short* __restrict__ dst,
                                         const unsigned short* __restrict__ src) {
    for (int idx = threadIdx.x; idx < 64 * 8; idx += 256) {
        const int key = idx >> 3, c = (idx & 7) * 8;
        union { uint4 q; unsigned short s[8]; } t;
        t.q = *(const uint4*)(src + key * 64 + c);
#pragma unroll
        for (int j = 0; j < 8; ++j) dst[(c + j) * 64 + key] = t.s[j];
    }
}

// ---------- shared GEMM mainloop ----------
// 128x128 block tile, BK=64, 8 waves (2x4), wave tile 64x32.
// LDS double-buffered (2 x 16KB per operand): async-stage slab k+1 while the
// matrix pipe chews slab k; one barrier + one asynccnt wait per K-step.
__device__ __forceinline__ void gemm_mainloop(const unsigned short* __restrict__ Ag,
                                              const unsigned short* __restrict__ Btg,
                                              int Kdim,
                                              unsigned short* lA, unsigned short* lB,
                                              v8f acc[4][2]) {
    const int wid = threadIdx.x >> 5;
    const int wm  = wid >> 2, wn = wid & 3;
    const int BUF = 128 * 64;

    stage_tile64(lA, Ag, 128, Kdim);
    stage_tile64(lB, Btg, 128, Kdim);
    stage_wait();
    __syncthreads();

    int p = 0;
    for (int k0 = 0; k0 < Kdim; k0 += 64) {
        const unsigned short* curA = lA + p * BUF;
        const unsigned short* curB = lB + p * BUF;
        if (k0 + 64 < Kdim) {  // prefetch next slab into the other buffer
            stage_tile64(lA + (p ^ 1) * BUF, Ag + k0 + 64, 128, Kdim);
            stage_tile64(lB + (p ^ 1) * BUF, Btg + k0 + 64, 128, Kdim);
        }
#pragma unroll
        for (int ks = 0; ks < 64; ks += 32) {
            v16bf af[4], bf[2];
#pragma unroll
            for (int fm = 0; fm < 4; ++fm) af[fm] = load_frag(curA, wm * 64 + fm * 16, ks, 64);
#pragma unroll
            for (int fn = 0; fn < 2; ++fn) bf[fn] = load_frag(curB, wn * 32 + fn * 16, ks, 64);
#pragma unroll
            for (int fm = 0; fm < 4; ++fm)
#pragma unroll
                for (int fn = 0; fn < 2; ++fn)
                    acc[fm][fn] = wmma_bf16(af[fm], bf[fn], acc[fm][fn]);
        }
        stage_wait();
        __syncthreads();
        p ^= 1;
    }
}

// ---------- conversion kernels ----------
__global__ __launch_bounds__(256) void cvt_copy_bf16(const float* __restrict__ in,
                                                     unsigned short* __restrict__ out, int n) {
    int i = blockIdx.x * 256 + threadIdx.x;
    if (i < n) out[i] = f2bf(in[i]);
}

__global__ __launch_bounds__(256) void cvt_transpose_bf16(const float* __restrict__ in, // [K][N]
                                                          unsigned short* __restrict__ out, // [N][K]
                                                          int K, int N) {
    int i = blockIdx.x * 256 + threadIdx.x;
    if (i < K * N) {
        int k = i / N, n = i - k * N;
        out[(size_t)n * K + k] = f2bf(in[i]);
    }
}

// ---------- QKV GEMM: [8192,1024] x [1024,3072] + bias -> per-head Q/K/V bf16 ----------
__global__ __launch_bounds__(256) void qkv_gemm(const unsigned short* __restrict__ Xb,
                                                const unsigned short* __restrict__ Wt, // [3072][1024]
                                                const float* __restrict__ bias,
                                                unsigned short* __restrict__ Qb,
                                                unsigned short* __restrict__ Kb,
                                                unsigned short* __restrict__ Vb) {
    __shared__ __align__(16) unsigned short lA[2 * 128 * 64];
    __shared__ __align__(16) unsigned short lB[2 * 128 * 64];
    v8f acc[4][2] = {};
    const int Kdim = 1024;
    gemm_mainloop(Xb + (size_t)blockIdx.y * 128 * Kdim,
                  Wt + (size_t)blockIdx.x * 128 * Kdim, Kdim, lA, lB, acc);

    const int lane = threadIdx.x & 31, half = lane >> 4;
    const int wid = threadIdx.x >> 5, wm = wid >> 2, wn = wid & 3;
#pragma unroll
    for (int fm = 0; fm < 4; ++fm)
#pragma unroll
        for (int fn = 0; fn < 2; ++fn)
#pragma unroll
            for (int r = 0; r < 8; ++r) {
                const int m = blockIdx.y * 128 + wm * 64 + fm * 16 + r + 8 * half;
                const int n = blockIdx.x * 128 + wn * 32 + fn * 16 + (lane & 15);
                float val = acc[fm][fn][r] + bias[n];
                const int bb = m >> 11, s = m & 2047;
                const int sec = n >> 10, col = n & 1023, hh = col >> 6, dd = col & 63;
                const size_t idx = (((size_t)(bb * 16 + hh) * 2048 + s) * 64 + dd);
                if (sec == 0)      Qb[idx] = f2bf(val * 0.125f);  // fold 1/sqrt(64) into Q
                else if (sec == 1) Kb[idx] = f2bf(val);
                else               Vb[idx] = f2bf(val);
            }
}

// ---------- flash attention: block = (b,h, 128-query tile), 8 waves x 16 rows ----------
__global__ __launch_bounds__(256) void flash_attn(const unsigned short* __restrict__ Qb,
                                                  const unsigned short* __restrict__ Kb,
                                                  const unsigned short* __restrict__ Vb,
                                                  unsigned short* __restrict__ Ab) {
    __shared__ __align__(16) unsigned short lQ[128 * 64];
    __shared__ __align__(16) unsigned short lK[64 * 64];
    __shared__ __align__(16) unsigned short lVt[64 * 64];
    __shared__ __align__(16) unsigned short lP[8][16 * 64];

    const int qt = blockIdx.x;         // query tile (16 of them)
    const int bh = blockIdx.y;         // b*16 + h
    const int b = bh >> 4, h = bh & 15;
    const size_t seq_base = (size_t)bh * 2048 * 64;
    const int wid = threadIdx.x >> 5, lane = threadIdx.x & 31, half = lane >> 4;

    stage_tile64(lQ, Qb + seq_base + (size_t)qt * 128 * 64, 128, 64);

    v8f o[4] = {};
    float mrow[8], lrow[8];
#pragma unroll
    for (int r = 0; r < 8; ++r) { mrow[r] = -3e38f; lrow[r] = 0.f; }

    const int ktmax = 2 * qt + 1;      // causal: keys <= qt*128+127
    for (int kt = 0; kt <= ktmax; ++kt) {
        __syncthreads();
        stage_tile64(lK, Kb + seq_base + (size_t)kt * 64 * 64, 64, 64);
        stage_vT(lVt, Vb + seq_base + (size_t)kt * 64 * 64);
        stage_wait();                  // Q (first iter) + K async writes complete
        __syncthreads();

        // scores S = Q K^T (already scaled via Q)
        v8f sc[4] = {};
        v16bf aq0 = load_frag(lQ, wid * 16, 0, 64);
        v16bf aq1 = load_frag(lQ, wid * 16, 32, 64);
#pragma unroll
        for (int fn = 0; fn < 4; ++fn) {
            v16bf bk0 = load_frag(lK, fn * 16, 0, 64);
            v16bf bk1 = load_frag(lK, fn * 16, 32, 64);
            sc[fn] = wmma_bf16(aq0, bk0, sc[fn]);
            sc[fn] = wmma_bf16(aq1, bk1, sc[fn]);
        }

        if (kt >= 2 * qt) {  // diagonal region: causal mask
            const int qbase = qt * 128 + wid * 16 + 8 * half;
            const int kbase = kt * 64 + (lane & 15);
#pragma unroll
            for (int fn = 0; fn < 4; ++fn)
#pragma unroll
                for (int r = 0; r < 8; ++r)
                    if (kbase + fn * 16 > qbase + r) sc[fn][r] = -1e9f;
        }

        // online softmax per row (row stats shared across a 16-lane half-wave)
#pragma unroll
        for (int r = 0; r < 8; ++r) {
            float mx = fmaxf(fmaxf(sc[0][r], sc[1][r]), fmaxf(sc[2][r], sc[3][r]));
#pragma unroll
            for (int ofs = 1; ofs < 16; ofs <<= 1) mx = fmaxf(mx, __shfl_xor(mx, ofs, 32));
            const float mnew  = fmaxf(mrow[r], mx);
            const float scale = __expf(mrow[r] - mnew);
            float rs = 0.f;
#pragma unroll
            for (int fn = 0; fn < 4; ++fn) {
                float p = __expf(sc[fn][r] - mnew);
                sc[fn][r] = p;
                rs += p;
            }
#pragma unroll
            for (int ofs = 1; ofs < 16; ofs <<= 1) rs += __shfl_xor(rs, ofs, 32);
            lrow[r] = lrow[r] * scale + rs;
            mrow[r] = mnew;
#pragma unroll
            for (int fq = 0; fq < 4; ++fq) o[fq][r] *= scale;
        }

        // P (D-layout f32) -> LDS bf16 in A-operand orientation
        unsigned short* myP = lP[wid];
#pragma unroll
        for (int fn = 0; fn < 4; ++fn)
#pragma unroll
            for (int r = 0; r < 8; ++r)
                myP[(r + 8 * half) * 64 + fn * 16 + (lane & 15)] = f2bf(sc[fn][r]);
        __syncthreads();

        // O += P @ V
#pragma unroll
        for (int kf = 0; kf < 2; ++kf) {
            v16bf ap = load_frag(myP, 0, kf * 32, 64);
#pragma unroll
            for (int fq = 0; fq < 4; ++fq) {
                v16bf bv = load_frag(lVt, fq * 16, kf * 32, 64);
                o[fq] = wmma_bf16(ap, bv, o[fq]);
            }
        }
    }

    // epilogue: O / l, merge heads -> Ab[B,S,H,hd] (== row-major [8192,1024])
#pragma unroll
    for (int fq = 0; fq < 4; ++fq)
#pragma unroll
        for (int r = 0; r < 8; ++r) {
            const float val = o[fq][r] / lrow[r];
            const int s = qt * 128 + wid * 16 + r + 8 * half;
            const size_t idx = ((size_t)(b * 2048 + s) * 16 + h) * 64 + fq * 16 + (lane & 15);
            Ab[idx] = f2bf(val);
        }
}

// ---------- projection GEMM: [8192,1024] x [1024,1024] + bias -> f32 out ----------
__global__ __launch_bounds__(256) void proj_gemm(const unsigned short* __restrict__ Ag,
                                                 const unsigned short* __restrict__ Wt, // [1024][1024]
                                                 const float* __restrict__ bias,
                                                 float* __restrict__ out) {
    __shared__ __align__(16) unsigned short lA[2 * 128 * 64];
    __shared__ __align__(16) unsigned short lB[2 * 128 * 64];
    v8f acc[4][2] = {};
    const int Kdim = 1024;
    gemm_mainloop(Ag + (size_t)blockIdx.y * 128 * Kdim,
                  Wt + (size_t)blockIdx.x * 128 * Kdim, Kdim, lA, lB, acc);

    const int lane = threadIdx.x & 31, half = lane >> 4;
    const int wid = threadIdx.x >> 5, wm = wid >> 2, wn = wid & 3;
#pragma unroll
    for (int fm = 0; fm < 4; ++fm)
#pragma unroll
        for (int fn = 0; fn < 2; ++fn)
#pragma unroll
            for (int r = 0; r < 8; ++r) {
                const int m = blockIdx.y * 128 + wm * 64 + fm * 16 + r + 8 * half;
                const int n = blockIdx.x * 128 + wn * 32 + fn * 16 + (lane & 15);
                out[(size_t)m * 1024 + n] = acc[fm][fn][r] + bias[n];
            }
}

// ---------- host launch ----------
extern "C" void kernel_launch(void* const* d_in, const int* in_sizes, int n_in,
                              void* d_out, int out_size, void* d_ws, size_t ws_size,
                              hipStream_t stream) {
    const float* x      = (const float*)d_in[0];  // [4,2048,1024]
    const float* W_attn = (const float*)d_in[1];  // [1024,3072]
    const float* b_attn = (const float*)d_in[2];  // [3072]
    const float* W_proj = (const float*)d_in[3];  // [1024,1024]
    const float* b_proj = (const float*)d_in[4];  // [1024]
    float* out = (float*)d_out;                   // [4,2048,1024]

    char* ws = (char*)d_ws;
    const size_t SZ16 = 8192ull * 1024 * 2;       // 16 MiB (bf16 [8192,1024])
    unsigned short* xb    = (unsigned short*)(ws);
    unsigned short* WatT  = (unsigned short*)(ws + SZ16);
    unsigned short* WprT  = (unsigned short*)(ws + SZ16 + 3072ull * 1024 * 2);
    unsigned short* Qb    = (unsigned short*)(ws + SZ16 + 3072ull * 1024 * 2 + 1024ull * 1024 * 2);
    unsigned short* Kb    = Qb + 8192ull * 1024;
    unsigned short* Vb    = Kb + 8192ull * 1024;
    unsigned short* Ab    = Vb + 8192ull * 1024;

    // 1) convert to bf16 (weights transposed to [N][K] for B-operand staging)
    cvt_copy_bf16<<<(8192 * 1024) / 256, 256, 0, stream>>>(x, xb, 8192 * 1024);
    cvt_transpose_bf16<<<(1024 * 3072) / 256, 256, 0, stream>>>(W_attn, WatT, 1024, 3072);
    cvt_transpose_bf16<<<(1024 * 1024) / 256, 256, 0, stream>>>(W_proj, WprT, 1024, 1024);

    // 2) QKV GEMM -> per-head Q/K/V (Q pre-scaled by 1/sqrt(hd))
    qkv_gemm<<<dim3(3072 / 128, 8192 / 128), 256, 0, stream>>>(xb, WatT, b_attn, Qb, Kb, Vb);

    // 3) causal flash attention -> Ab (merged heads, bf16 [8192,1024])
    flash_attn<<<dim3(2048 / 128, 4 * 16), 256, 0, stream>>>(Qb, Kb, Vb, Ab);

    // 4) output projection -> f32
    proj_gemm<<<dim3(1024 / 128, 8192 / 128), 256, 0, stream>>>(Ab, WprT, b_proj, out);

    (void)in_sizes; (void)n_in; (void)out_size; (void)ws_size;
}
